// dMaSIFSearchEmbed_54760833024291
// MI455X (gfx1250) — matrix operand order repaired and checked by reading, not compiled, not run on hardware
//
#include <hip/hip_runtime.h>
#include <cstdint>

#define NP      2048
#define CIN     16
#define HIDN    16
#define NCUTS   8
#define SLOPE   0.2f
#define RADIUS_F 9.0f
#define CONV_WAVES 8

typedef __bf16 bf16;
typedef bf16  v16bf __attribute__((ext_vector_type(16)));
typedef float v8f   __attribute__((ext_vector_type(8)));

__device__ __forceinline__ float leaky(float x) { return fmaxf(x, SLOPE * x); }

// ---------------------------------------------------------------------------
// Kernel: per-point prep (shared by both branches)
//   p1 = xyz/R, p2 = xyz/(sqrt(2)R), tangent vectors uv (2x3 per point)
// ---------------------------------------------------------------------------
__global__ void k_prep(const float* __restrict__ xyz, const float* __restrict__ nrm,
                       float* __restrict__ p1, float* __restrict__ p2,
                       float* __restrict__ uvb) {
  int i = blockIdx.x * blockDim.x + threadIdx.x;
  if (i >= NP) return;
  float x = xyz[3*i+0], y = xyz[3*i+1], z = xyz[3*i+2];
  const float inv1 = 1.0f / RADIUS_F;
  const float inv2 = 1.0f / (1.41421356237f * RADIUS_F);
  p1[4*i+0] = x*inv1; p1[4*i+1] = y*inv1; p1[4*i+2] = z*inv1; p1[4*i+3] = 0.f;
  p2[4*i+0] = x*inv2; p2[4*i+1] = y*inv2; p2[4*i+2] = z*inv2; p2[4*i+3] = 0.f;
  float nx = nrm[3*i+0], ny = nrm[3*i+1], nz = nrm[3*i+2];
  float s = (nz >= 0.f) ? 1.f : -1.f;
  float a = -1.f / (s + nz);
  float b = nx * ny * a;
  uvb[8*i+0] = 1.f + s*nx*nx*a;
  uvb[8*i+1] = s*b;
  uvb[8*i+2] = -s*nx;
  uvb[8*i+3] = 0.f;
  uvb[8*i+4] = b;
  uvb[8*i+5] = s + ny*ny*a;
  uvb[8*i+6] = -ny;
  uvb[8*i+7] = 0.f;
}

// ---------------------------------------------------------------------------
// Kernel: weight MLP  w = lin2(leaky(lin1(feat)))  (scalar per point)
// ---------------------------------------------------------------------------
__global__ void k_wmlp(const float* __restrict__ feats,
                       const float* __restrict__ w1, const float* __restrict__ b1,
                       const float* __restrict__ w2, const float* __restrict__ b2,
                       float* __restrict__ wv) {
  int i = blockIdx.x * blockDim.x + threadIdx.x;
  if (i >= NP) return;
  float x[CIN];
#pragma unroll
  for (int c = 0; c < CIN; c++) x[c] = feats[CIN*i + c];
  float h[HIDN];
#pragma unroll
  for (int r = 0; r < HIDN; r++) {
    float a = b1[r];
#pragma unroll
    for (int c = 0; c < CIN; c++) a += w1[r*CIN + c] * x[c];
    h[r] = leaky(a);
  }
  float o = b2[0];
#pragma unroll
  for (int c = 0; c < HIDN; c++) o += w2[c] * h[c];
  wv[i] = o;
}

// ---------------------------------------------------------------------------
// Kernel: mesh N^2 pass.  One wave per i.  m[i] = sum_j S_ij p1_j, t[i]=sum S.
// ---------------------------------------------------------------------------
__global__ void k_mesh(const float* __restrict__ p1, const float* __restrict__ nrm,
                       const float* __restrict__ wv, float* __restrict__ mt) {
  const int lane = threadIdx.x & 31;
  const int i = blockIdx.x * (blockDim.x >> 5) + (threadIdx.x >> 5);
  float pix = p1[4*i+0], piy = p1[4*i+1], piz = p1[4*i+2];
  float nix = nrm[3*i+0], niy = nrm[3*i+1], niz = nrm[3*i+2];
  float mx = 0.f, my = 0.f, mz = 0.f, tt = 0.f;
  for (int j = lane; j < NP; j += 32) {
    float pjx = p1[4*j+0], pjy = p1[4*j+1], pjz = p1[4*j+2];
    float njx = nrm[3*j+0], njy = nrm[3*j+1], njz = nrm[3*j+2];
    float dx = pix - pjx, dy = piy - pjy, dz = piz - pjz;
    float f2 = 2.f - (nix*njx + niy*njy + niz*njz);
    float S = __expf(-(dx*dx + dy*dy + dz*dz) * f2 * f2) * wv[j];
    mx += S * pjx; my += S * pjy; mz += S * pjz; tt += S;
  }
#pragma unroll
  for (int off = 16; off > 0; off >>= 1) {
    mx += __shfl_xor(mx, off, 32);
    my += __shfl_xor(my, off, 32);
    mz += __shfl_xor(mz, off, 32);
    tt += __shfl_xor(tt, off, 32);
  }
  if (lane == 0) { mt[4*i+0] = mx; mt[4*i+1] = my; mt[4*i+2] = mz; mt[4*i+3] = tt; }
}

// ---------------------------------------------------------------------------
// Kernel: build local frames nuv (3 rows x 3 per point, contiguous, stride 12)
// ---------------------------------------------------------------------------
__global__ void k_frames(const float* __restrict__ p1, const float* __restrict__ nrm,
                         const float* __restrict__ uvb, const float* __restrict__ mt,
                         float* __restrict__ nuv) {
  int i = blockIdx.x * blockDim.x + threadIdx.x;
  if (i >= NP) return;
  float u0x = uvb[8*i+0], u0y = uvb[8*i+1], u0z = uvb[8*i+2];
  float u1x = uvb[8*i+4], u1y = uvb[8*i+5], u1z = uvb[8*i+6];
  float mx = mt[4*i+0], my = mt[4*i+1], mz = mt[4*i+2], t = mt[4*i+3];
  float px = p1[4*i+0], py = p1[4*i+1], pz = p1[4*i+2];
  float ov0 = (u0x*mx + u0y*my + u0z*mz) - t*(u0x*px + u0y*py + u0z*pz) + 1e-5f;
  float ov1 = (u1x*mx + u1y*my + u1z*mz) - t*(u1x*px + u1y*py + u1z*pz) + 1e-5f;
  float inv = rsqrtf(ov0*ov0 + ov1*ov1);
  float ex = ov0 * inv, ey = ov1 * inv;
  nuv[12*i+0] = nrm[3*i+0];
  nuv[12*i+1] = nrm[3*i+1];
  nuv[12*i+2] = nrm[3*i+2];
  nuv[12*i+3] =  ex*u0x + ey*u1x;
  nuv[12*i+4] =  ex*u0y + ey*u1y;
  nuv[12*i+5] =  ex*u0z + ey*u1z;
  nuv[12*i+6] = -ey*u0x + ex*u1x;
  nuv[12*i+7] = -ey*u0y + ex*u1y;
  nuv[12*i+8] = -ey*u0z + ex*u1z;
  nuv[12*i+9] = 0.f; nuv[12*i+10] = 0.f; nuv[12*i+11] = 0.f;
}

// ---------------------------------------------------------------------------
// Kernel: two leaky-linear layers (16->16->16) + group-norm stats (atomics)
// ---------------------------------------------------------------------------
__global__ void k_mlp2_stats(const float* __restrict__ xin,
                             const float* __restrict__ w1, const float* __restrict__ b1,
                             const float* __restrict__ w2, const float* __restrict__ b2,
                             float* __restrict__ yout, float* __restrict__ gst) {
  int i = blockIdx.x * blockDim.x + threadIdx.x;
  if (i >= NP) return;
  float x[HIDN], h1[HIDN], h2[HIDN];
#pragma unroll
  for (int c = 0; c < HIDN; c++) x[c] = xin[HIDN*i + c];
#pragma unroll
  for (int r = 0; r < HIDN; r++) {
    float a = b1[r];
#pragma unroll
    for (int c = 0; c < HIDN; c++) a += w1[r*HIDN + c] * x[c];
    h1[r] = leaky(a);
  }
#pragma unroll
  for (int r = 0; r < HIDN; r++) {
    float a = b2[r];
#pragma unroll
    for (int c = 0; c < HIDN; c++) a += w2[r*HIDN + c] * h1[c];
    h2[r] = leaky(a);
    yout[HIDN*i + r] = h2[r];
  }
#pragma unroll
  for (int g = 0; g < 4; g++) {
    float s = 0.f, ss = 0.f;
#pragma unroll
    for (int k = 0; k < 4; k++) { float v = h2[4*g + k]; s += v; ss += v*v; }
    atomicAdd(&gst[2*g + 0], s);
    atomicAdd(&gst[2*g + 1], ss);
  }
}

// ---------------------------------------------------------------------------
// Kernel: apply group norm, write f as bf16 transposed (ftr[h][j]) for WMMA B
// ---------------------------------------------------------------------------
__global__ void k_gn_to_bf16(const float* __restrict__ xin,
                             const float* __restrict__ gamma, const float* __restrict__ beta,
                             const float* __restrict__ gst, bf16* __restrict__ ftr) {
  int i = blockIdx.x * blockDim.x + threadIdx.x;
  if (i >= NP) return;
  const float invn = 1.0f / (4.0f * NP);
#pragma unroll
  for (int c = 0; c < HIDN; c++) {
    int g = c >> 2;
    float mu  = gst[2*g + 0] * invn;
    float var = gst[2*g + 1] * invn - mu*mu;
    float xn  = (xin[HIDN*i + c] - mu) * rsqrtf(var + 1e-5f);
    ftr[(size_t)c * NP + i] = (bf16)(xn * gamma[c] + beta[c]);
  }
}

// ---------------------------------------------------------------------------
// Kernel: the heavy N^2 conv pass, WMMA bf16 16x16x32 per 32-j chunk.
// One wave per output point i.  G rows 0..7 = cuts, row 8 = window sum (s).
// ---------------------------------------------------------------------------
__global__ void k_conv(const float* __restrict__ p2g, const float* __restrict__ nrm,
                       const float* __restrict__ nuv, const bf16* __restrict__ ftr,
                       const float* __restrict__ cw1, const float* __restrict__ cb1,
                       const float* __restrict__ cw2, const float* __restrict__ cb2,
                       float* __restrict__ out0) {
  extern __shared__ char smem[];
  bf16* ftile  = (bf16*)smem;                                    // 16 x NP bf16
  bf16* astage = (bf16*)(smem + (size_t)16 * NP * sizeof(bf16)); // waves x 16 x 32

  // cooperative copy f^T (bf16) into LDS (global_load_b128 + ds_store_b128)
  {
    const float4* src = (const float4*)ftr;
    float4* dst = (float4*)ftile;
    const int nchunk = (16 * NP) / 8;
    for (int idx = threadIdx.x; idx < nchunk; idx += blockDim.x) dst[idx] = src[idx];
  }
  __syncthreads();

  const int lane = threadIdx.x & 31;
  const int wv   = threadIdx.x >> 5;
  const int i    = blockIdx.x * CONV_WAVES + wv;   // grid sized exactly: always valid
  bf16* arow = astage + wv * (16 * 32);            // wave-private staging slab

  const float pix = p2g[4*i+0], piy = p2g[4*i+1], piz = p2g[4*i+2];
  const float nix = nrm[3*i+0], niy = nrm[3*i+1], niz = nrm[3*i+2];
  float fr[9];
#pragma unroll
  for (int k = 0; k < 9; k++) fr[k] = nuv[12*i + k];

  float w1r[3*NCUTS], b1r[NCUTS];
#pragma unroll
  for (int k = 0; k < 3*NCUTS; k++) w1r[k] = cw1[k];
#pragma unroll
  for (int c = 0; c < NCUTS; c++) b1r[c] = cb1[c];

  v8f acc;
#pragma unroll
  for (int e = 0; e < 8; e++) acc[e] = 0.f;

  const int mrow = lane & 15;
  const int hi   = lane >> 4;

  for (int jb = 0; jb < NP; jb += 32) {
    const int j = jb + lane;
    float pjx = p2g[4*j+0], pjy = p2g[4*j+1], pjz = p2g[4*j+2];
    float njx = nrm[3*j+0], njy = nrm[3*j+1], njz = nrm[3*j+2];
    float dx = pjx - pix, dy = pjy - piy, dz = pjz - piz;   // diff = p_j - p_i
    float f2 = 2.f - (nix*njx + niy*njy + niz*njz);
    float win = __expf(-(dx*dx + dy*dy + dz*dz) * f2 * f2);
    float X0 = fr[0]*dx + fr[1]*dy + fr[2]*dz;
    float X1 = fr[3]*dx + fr[4]*dy + fr[5]*dz;
    float X2 = fr[6]*dx + fr[7]*dy + fr[8]*dz;

    // stage A[c][j] = win * relu(w1.X + b1), A[8][j] = win  (bf16, k-major rows)
#pragma unroll
    for (int c = 0; c < NCUTS; c++) {
      float r = fmaxf(w1r[3*c+0]*X0 + w1r[3*c+1]*X1 + w1r[3*c+2]*X2 + b1r[c], 0.f);
      arow[c*32 + lane] = (bf16)(win * r);
    }
    arow[8*32 + lane] = (bf16)win;

    // DS ops are in-order per wave; this is a compiler barrier + hard wait
    asm volatile("s_wait_dscnt 0x0" ::: "memory");

    // A fragment: lane holds row m=lane%16; elems 0..7 = k base, 8..15 = k base+16
    v16bf A;
#pragma unroll
    for (int e = 0; e < 16; e++) A[e] = (bf16)0.f;
    if (mrow < 9) {
      const bf16* ab = arow + mrow*32 + hi*8;
      ((float4*)&A)[0] = *(const float4*)(ab);
      ((float4*)&A)[1] = *(const float4*)(ab + 16);
    }
    // B fragment: lane holds col n=lane%16; elems 0..7 = k base, 8..15 = k base+16
    v16bf B;
    const bf16* bb = ftile + (size_t)mrow * NP + jb + hi*8;
    ((float4*)&B)[0] = *(const float4*)(bb);
    ((float4*)&B)[1] = *(const float4*)(bb + 16);

    asm volatile("" ::: "memory");  // keep loads(t) ordered before stores(t+1)

    acc = __builtin_amdgcn_wmma_f32_16x16x32_bf16(
        /*neg_a=*/false, A, /*neg_b=*/false, B,
        /*c_mod=*/(short)0, acc, /*reuse_a=*/false, /*reuse_b=*/false);
  }

  // C layout: lanes 0-15 hold rows 0..7 (n=lane); lanes 16-31 hold rows 8..15.
  // Row 8 (the window-sum s) lives in acc[0] of lanes 16..31.
  float sv = __shfl(acc[0], mrow + 16, 32);
  if (lane < 16) {
    float o = cb2[lane] * sv;
#pragma unroll
    for (int c = 0; c < NCUTS; c++) o += cw2[lane*NCUTS + c] * acc[c];
    out0[i*HIDN + lane] = o;
  }
}

// ---------------------------------------------------------------------------
// Kernel: final GN apply + ll MLP + residual lt linear
// ---------------------------------------------------------------------------
__global__ void k_final(const float* __restrict__ ypre,
                        const float* __restrict__ gamma, const float* __restrict__ beta,
                        const float* __restrict__ gst,
                        const float* __restrict__ llw1, const float* __restrict__ llb1,
                        const float* __restrict__ llw2, const float* __restrict__ llb2,
                        const float* __restrict__ ltw, const float* __restrict__ ltb,
                        const float* __restrict__ feats, float* __restrict__ eout) {
  int i = blockIdx.x * blockDim.x + threadIdx.x;
  if (i >= NP) return;
  const float invn = 1.0f / (4.0f * NP);
  float xg[HIDN];
#pragma unroll
  for (int c = 0; c < HIDN; c++) {
    int g = c >> 2;
    float mu  = gst[2*g + 0] * invn;
    float var = gst[2*g + 1] * invn - mu*mu;
    float xn  = (ypre[HIDN*i + c] - mu) * rsqrtf(var + 1e-5f);
    xg[c] = xn * gamma[c] + beta[c];
  }
  float h1[HIDN];
#pragma unroll
  for (int r = 0; r < HIDN; r++) {
    float a = llb1[r];
#pragma unroll
    for (int c = 0; c < HIDN; c++) a += llw1[r*HIDN + c] * xg[c];
    h1[r] = fmaxf(a, 0.f);
  }
  float fin[CIN];
#pragma unroll
  for (int c = 0; c < CIN; c++) fin[c] = feats[CIN*i + c];
#pragma unroll
  for (int h = 0; h < HIDN; h++) {
    float a = llb2[h];
#pragma unroll
    for (int c = 0; c < HIDN; c++) a += llw2[h*HIDN + c] * h1[c];
    float base = ltb[h];
#pragma unroll
    for (int c = 0; c < CIN; c++) base += ltw[h*CIN + c] * fin[c];
    eout[i*HIDN + h] = base + a;
  }
}

// ---------------------------------------------------------------------------
// Host launch
// ---------------------------------------------------------------------------
enum {
  P_OS_W1 = 0, P_OS_B1, P_OS_W2, P_OS_B2,
  P_IN_W1, P_IN_B1, P_IN_W2, P_IN_B2,
  P_GN_IN_G, P_GN_IN_B,
  P_CONV_W1, P_CONV_B1, P_CONV_W2, P_CONV_B2,
  P_OUT_W1, P_OUT_B1, P_OUT_W2, P_OUT_B2,
  P_GN_OUT_G, P_GN_OUT_B,
  P_LL_W1, P_LL_B1, P_LL_W2, P_LL_B2,
  P_LT_W, P_LT_B, P_COUNT
};

extern "C" void kernel_launch(void* const* d_in, const int* in_sizes, int n_in,
                              void* d_out, int out_size, void* d_ws, size_t ws_size,
                              hipStream_t stream) {
  (void)in_sizes; (void)out_size; (void)ws_size;
  const float* xyz   = (const float*)d_in[0];
  const float* nrm   = (const float*)d_in[1];
  const float* feats = (const float*)d_in[2];
  float* ws = (float*)d_ws;

  // workspace layout (floats): ~1.3 MB total
  float* gst  = ws;               // 64 floats: [branch][stage][group][sum,sumsq]
  float* p1   = ws + 64;          // NP*4
  float* p2   = p1 + 4*NP;        // NP*4
  float* uvb  = p2 + 4*NP;        // NP*8
  float* bb0  = uvb + 8*NP;       // per-branch blocks of 74*NP floats

  hipMemsetAsync(gst, 0, 64 * sizeof(float), stream);
  k_prep<<<NP/256, 256, 0, stream>>>(xyz, nrm, p1, p2, uvb);

  for (int b = 0; b < 2; b++) {
    const float* P[P_COUNT];
    for (int k = 0; k < P_COUNT; k++) P[k] = (const float*)d_in[3 + b*P_COUNT + k];

    float* base = bb0 + (size_t)b * 74 * NP;
    float* wv   = base;             // NP
    float* mt   = wv + NP;          // NP*4
    float* nuv  = mt + 4*NP;        // NP*12
    float* fpre = nuv + 12*NP;      // NP*16
    float* out0 = fpre + 16*NP;     // NP*16
    float* ypre = out0 + 16*NP;     // NP*16
    bf16*  ftr  = (bf16*)(ypre + 16*NP); // 16*NP bf16 == NP*8 floats
    float* gstA = gst + b*16;
    float* gstB = gstA + 8;

    k_wmlp<<<NP/256, 256, 0, stream>>>(feats, P[P_OS_W1], P[P_OS_B1],
                                       P[P_OS_W2], P[P_OS_B2], wv);
    k_mesh<<<NP/8, 256, 0, stream>>>(p1, nrm, wv, mt);
    k_frames<<<NP/256, 256, 0, stream>>>(p1, nrm, uvb, mt, nuv);
    k_mlp2_stats<<<NP/256, 256, 0, stream>>>(feats, P[P_IN_W1], P[P_IN_B1],
                                             P[P_IN_W2], P[P_IN_B2], fpre, gstA);
    k_gn_to_bf16<<<NP/256, 256, 0, stream>>>(fpre, P[P_GN_IN_G], P[P_GN_IN_B],
                                             gstA, ftr);
    size_t smem = (size_t)16 * NP * sizeof(bf16)
                + (size_t)CONV_WAVES * 16 * 32 * sizeof(bf16);
    k_conv<<<NP/CONV_WAVES, CONV_WAVES*32, smem, stream>>>(
        p2, nrm, nuv, ftr, P[P_CONV_W1], P[P_CONV_B1],
        P[P_CONV_W2], P[P_CONV_B2], out0);
    k_mlp2_stats<<<NP/256, 256, 0, stream>>>(out0, P[P_OUT_W1], P[P_OUT_B1],
                                             P[P_OUT_W2], P[P_OUT_B2], ypre, gstB);
    k_final<<<NP/256, 256, 0, stream>>>(ypre, P[P_GN_OUT_G], P[P_GN_OUT_B], gstB,
                                        P[P_LL_W1], P[P_LL_B1], P[P_LL_W2], P[P_LL_B2],
                                        P[P_LT_W], P[P_LT_B], feats,
                                        (float*)d_out + (size_t)b * NP * HIDN);
  }
}
